// MythosAttention_75488345195160
// MI455X (gfx1250) — matrix operand order repaired and checked by reading, not compiled, not run on hardware
//
#include <hip/hip_runtime.h>

#define BATCH 2
#define SEQ   2048
#define HID   2048
#define NHEAD 16
#define HDIM  128

typedef __attribute__((ext_vector_type(16))) __bf16 v16bf;
typedef __attribute__((ext_vector_type(8)))  float  v8f;

union Frag16 {
  v16bf v;
  uint4 q[2];
};

__device__ __forceinline__ unsigned short f32_to_bf16(float f) {
  unsigned int u = __float_as_uint(f);
  u += 0x7FFFu + ((u >> 16) & 1u);   // round-to-nearest-even
  return (unsigned short)(u >> 16);
}

// ---------------------------------------------------------------------------
// C = A (MxK, f32, row-major) * B^T   with B stored (NxK, f32, row-major).
// Converts both operands to bf16 while staging into LDS, accumulates f32 via
// v_wmma_f32_16x16x32_bf16. Block = 256 threads (8 waves), tile 128x128, K=32.
// ---------------------------------------------------------------------------
template <bool OUT_BF16>
__global__ __launch_bounds__(256)
void gemm_nt_wmma(const float* __restrict__ A, const float* __restrict__ Bm,
                  void* __restrict__ Cout, int M, int N, int K) {
  __shared__ unsigned short As[128][40];   // 32-K slab + 8-short pad (16B align kept)
  __shared__ unsigned short Bs[128][40];

  const int tid  = threadIdx.x;
  const int wid  = tid >> 5;
  const int lane = tid & 31;
  const int hi   = lane >> 4;
  const int ln16 = lane & 15;
  const int bm   = blockIdx.y * 128;
  const int bn   = blockIdx.x * 128;
  const int wm   = (wid & 3) * 32;   // 4 waves along M
  const int wn   = (wid >> 2) * 64;  // 2 waves along N

  v8f acc[2][4] = {};

  for (int k0 = 0; k0 < K; k0 += 32) {
    // stage 128x32 f32 -> bf16 for A and B (4096 elems each, float4 per thread)
#pragma unroll
    for (int it = 0; it < 4; ++it) {
      int idx = (it * 256 + tid) * 4;
      int r = idx >> 5;
      int c = idx & 31;
      float4 a4 = *(const float4*)(A + (size_t)(bm + r) * K + (k0 + c));
      unsigned short ta[4] = {f32_to_bf16(a4.x), f32_to_bf16(a4.y),
                              f32_to_bf16(a4.z), f32_to_bf16(a4.w)};
      *(uint2*)(&As[r][c]) = *(const uint2*)ta;
      float4 b4 = *(const float4*)(Bm + (size_t)(bn + r) * K + (k0 + c));
      unsigned short tb[4] = {f32_to_bf16(b4.x), f32_to_bf16(b4.y),
                              f32_to_bf16(b4.z), f32_to_bf16(b4.w)};
      *(uint2*)(&Bs[r][c]) = *(const uint2*)tb;
    }
    if (k0 + 32 < K) {  // warm next K-slab in GL2 (global_prefetch_b8)
      __builtin_prefetch(A + (size_t)(bm + (tid >> 1)) * K + k0 + 32, 0, 3);
      __builtin_prefetch(Bm + (size_t)(bn + (tid >> 1)) * K + k0 + 32, 0, 3);
    }
    __syncthreads();

    // A frag 16x32: lane<16 holds K {0-7,16-23}, lane>=16 holds K {8-15,24-31}
    Frag16 af[2], bf[4];
#pragma unroll
    for (int i = 0; i < 2; ++i) {
      const unsigned short* p = &As[wm + i * 16 + ln16][hi ? 8 : 0];
      af[i].q[0] = *(const uint4*)(p);
      af[i].q[1] = *(const uint4*)(p + 16);
    }
    // B frag 32x16: lane n%16 = column, lane<16 holds K {0-7,16-23}, lane>=16 holds K {8-15,24-31}
#pragma unroll
    for (int j = 0; j < 4; ++j) {
      const unsigned short* p = &Bs[wn + j * 16 + ln16][hi * 8];
      bf[j].q[0] = *(const uint4*)(p);
      bf[j].q[1] = *(const uint4*)(p + 16);
    }
#pragma unroll
    for (int i = 0; i < 2; ++i)
#pragma unroll
      for (int j = 0; j < 4; ++j)
        acc[i][j] = __builtin_amdgcn_wmma_f32_16x16x32_bf16(
            false, af[i].v, false, bf[j].v, (short)0, acc[i][j], false, false);
    __syncthreads();
  }

  // C layout: element e of vgpr -> row e + 8*hi, col = ln16
#pragma unroll
  for (int i = 0; i < 2; ++i)
#pragma unroll
    for (int j = 0; j < 4; ++j)
#pragma unroll
      for (int e = 0; e < 8; ++e) {
        int r = bm + wm + i * 16 + hi * 8 + e;
        int c = bn + wn + j * 16 + ln16;
        if (OUT_BF16)
          ((unsigned short*)Cout)[(size_t)r * N + c] = f32_to_bf16(acc[i][j][e]);
        else
          ((float*)Cout)[(size_t)r * N + c] = acc[i][j][e];
      }
}

// ---------------------------------------------------------------------------
// Flash attention: block = 4 waves = 64 queries of one (batch, head).
// K/V tiles (64 keys) staged in LDS (V transposed), online softmax with
// wave32 half-group shuffles, P via per-wave LDS scratch, all MMAs on WMMA.
// ---------------------------------------------------------------------------
__global__ __launch_bounds__(128)
void flash_attn_wmma(const unsigned short* __restrict__ Qb,
                     const unsigned short* __restrict__ Kb,
                     const unsigned short* __restrict__ Vb,
                     const float* __restrict__ mask,
                     float* __restrict__ ctx) {
  __shared__ unsigned short Ks[64][136];   // [key][dim]
  __shared__ unsigned short Vt[128][72];   // [dim][key] (transposed)
  __shared__ unsigned short Ps[4][16][72]; // per-wave P scratch [row][key]

  const int tid  = threadIdx.x;
  const int wid  = tid >> 5;
  const int lane = tid & 31;
  const int hi   = lane >> 4;
  const int ln16 = lane & 15;

  const int qblocks = SEQ / 64;
  const int qb   = blockIdx.x % qblocks;
  const int head = (blockIdx.x / qblocks) % NHEAD;
  const int b    = blockIdx.x / (qblocks * NHEAD);
  const int q0   = qb * 64 + wid * 16;

  const size_t rowbase = (size_t)b * SEQ;
  const float scale = 0.088388347648318447f;  // 1/sqrt(128)

  // Q fragments: 16 queries x 128 dims = 4 A-frags, loaded straight from global
  Frag16 aq[4];
  {
    const unsigned short* qp = Qb + (rowbase + q0 + ln16) * HID + head * HDIM;
#pragma unroll
    for (int f = 0; f < 4; ++f) {
      aq[f].q[0] = *(const uint4*)(qp + f * 32 + (hi ? 8 : 0));
      aq[f].q[1] = *(const uint4*)(qp + f * 32 + (hi ? 8 : 0) + 16);
    }
  }

  v8f acc_o[8] = {};
  float mrow[8], lrow[8];
#pragma unroll
  for (int e = 0; e < 8; ++e) { mrow[e] = -__builtin_inff(); lrow[e] = 0.0f; }

  for (int kt = 0; kt < SEQ; kt += 64) {
    __syncthreads();  // previous iteration done with K/V tiles
    // stage K tile: 64x128 bf16, row-major copy
#pragma unroll
    for (int it = 0; it < 4; ++it) {
      int idx = (it * 128 + tid) * 16;
      int r = idx >> 7;
      int c = idx & 127;
      const unsigned short* kp = Kb + (rowbase + kt + r) * HID + head * HDIM + c;
      *(uint4*)&Ks[r][c]     = *(const uint4*)(kp);
      *(uint4*)&Ks[r][c + 8] = *(const uint4*)(kp + 8);
    }
    // stage V tile transposed: Vt[d][key]
#pragma unroll
    for (int it = 0; it < 8; ++it) {
      int idx = (it * 128 + tid) * 8;
      int r = idx >> 7;
      int c = idx & 127;
      uint4 d4 = *(const uint4*)(Vb + (rowbase + kt + r) * HID + head * HDIM + c);
      unsigned short tmp[8];
      *(uint4*)tmp = d4;
#pragma unroll
      for (int j = 0; j < 8; ++j) Vt[c + j][r] = tmp[j];
    }
    __syncthreads();

    // S = Q K^T : 4 score tiles of 16x16, each = 4 chained WMMAs over d
    v8f sc[4];
#pragma unroll
    for (int ks = 0; ks < 4; ++ks) {
      v8f s = {};
#pragma unroll
      for (int kf = 0; kf < 4; ++kf) {
        Frag16 bk;
        const unsigned short* pb = &Ks[ks * 16 + ln16][kf * 32 + hi * 8];
        bk.q[0] = *(const uint4*)(pb);
        bk.q[1] = *(const uint4*)(pb + 16);
        s = __builtin_amdgcn_wmma_f32_16x16x32_bf16(
            false, aq[kf].v, false, bk.v, (short)0, s, false, false);
      }
      float madd = mask[(size_t)b * SEQ + kt + ks * 16 + ln16];
#pragma unroll
      for (int e = 0; e < 8; ++e) s[e] = s[e] * scale + madd;
      sc[ks] = s;
    }

    // online softmax: row = e + 8*hi; row spans 16 lanes of a half-wave
    float newm[8], alpha[8], rsum[8];
#pragma unroll
    for (int e = 0; e < 8; ++e) {
      float v = fmaxf(fmaxf(sc[0][e], sc[1][e]), fmaxf(sc[2][e], sc[3][e]));
#pragma unroll
      for (int off = 1; off < 16; off <<= 1) v = fmaxf(v, __shfl_xor(v, off, 32));
      newm[e]  = fmaxf(mrow[e], v);
      alpha[e] = __expf(mrow[e] - newm[e]);
      mrow[e]  = newm[e];
      rsum[e]  = 0.0f;
    }
#pragma unroll
    for (int ks = 0; ks < 4; ++ks)
#pragma unroll
      for (int e = 0; e < 8; ++e) {
        float p = __expf(sc[ks][e] - newm[e]);
        rsum[e] += p;
        Ps[wid][hi * 8 + e][ks * 16 + ln16] = f32_to_bf16(p);  // C-layout -> LDS
      }
#pragma unroll
    for (int e = 0; e < 8; ++e) {
      float v = rsum[e];
#pragma unroll
      for (int off = 1; off < 16; off <<= 1) v += __shfl_xor(v, off, 32);
      lrow[e] = lrow[e] * alpha[e] + v;
    }
#pragma unroll
    for (int dt = 0; dt < 8; ++dt)
#pragma unroll
      for (int e = 0; e < 8; ++e) acc_o[dt][e] *= alpha[e];

    __builtin_amdgcn_wave_barrier();  // same-wave DS ops are in-order; pin order

    // ctx += P (16x64) x V (64x128): 8 d-tiles x 2 key-chunks of K=32
#pragma unroll
    for (int dt = 0; dt < 8; ++dt)
#pragma unroll
      for (int kc = 0; kc < 2; ++kc) {
        Frag16 ap, bv;
        const unsigned short* pa = &Ps[wid][ln16][kc * 32 + (hi ? 8 : 0)];
        ap.q[0] = *(const uint4*)(pa);
        ap.q[1] = *(const uint4*)(pa + 16);
        const unsigned short* pv = &Vt[dt * 16 + ln16][kc * 32 + hi * 8];
        bv.q[0] = *(const uint4*)(pv);
        bv.q[1] = *(const uint4*)(pv + 16);
        acc_o[dt] = __builtin_amdgcn_wmma_f32_16x16x32_bf16(
            false, ap.v, false, bv.v, (short)0, acc_o[dt], false, false);
      }
  }

  // normalize and write ctx (f32, (b,s,head*128+d) layout)
#pragma unroll
  for (int e = 0; e < 8; ++e) {
    float inv = 1.0f / lrow[e];
    int r = hi * 8 + e;
#pragma unroll
    for (int dt = 0; dt < 8; ++dt)
      ctx[(rowbase + q0 + r) * HID + head * HDIM + dt * 16 + ln16] =
          acc_o[dt][e] * inv;
  }
}

// ---------------------------------------------------------------------------
extern "C" void kernel_launch(void* const* d_in, const int* in_sizes, int n_in,
                              void* d_out, int out_size, void* d_ws, size_t ws_size,
                              hipStream_t stream) {
  const float* X    = (const float*)d_in[0];
  const float* mask = (const float*)d_in[1];
  const float* Wq   = (const float*)d_in[2];
  const float* Wk   = (const float*)d_in[3];
  const float* Wv   = (const float*)d_in[4];
  const float* Wo   = (const float*)d_in[5];

  const int M = BATCH * SEQ;            // 4096 token rows
  const size_t mat = (size_t)M * HID;   // 8.39M elements

  // workspace: Q/K/V bf16 (3 x 16.8MB) + ctx f32 (33.6MB) = ~84MB
  unsigned short* Qb = (unsigned short*)d_ws;
  unsigned short* Kb = Qb + mat;
  unsigned short* Vb = Kb + mat;
  float*          Cx = (float*)(Vb + mat);

  dim3 grid(HID / 128, M / 128);  // 16 x 32 tiles

  gemm_nt_wmma<true ><<<grid, 256, 0, stream>>>(X,  Wq, Qb,   M, HID, HID);
  gemm_nt_wmma<true ><<<grid, 256, 0, stream>>>(X,  Wk, Kb,   M, HID, HID);
  gemm_nt_wmma<true ><<<grid, 256, 0, stream>>>(X,  Wv, Vb,   M, HID, HID);

  flash_attn_wmma<<<BATCH * NHEAD * (SEQ / 64), 128, 0, stream>>>(Qb, Kb, Vb, mask, Cx);

  gemm_nt_wmma<false><<<grid, 256, 0, stream>>>(Cx, Wo, d_out, M, HID, HID);
}
